// GCAttention_channel_81106162417831
// MI455X (gfx1250) — compile-verified, matrix operand order checked by reading, not run on hardware
//
#include <hip/hip_runtime.h>

#define B_   32
#define C_   512
#define H_   56
#define W_   56
#define S_   (H_ * W_)   // 3136
#define P_   64
#define TS_  448         // spatial chunk; 7 * 448 == 3136
#define NCH_ 7
#define EPS_ 1e-3f

typedef __attribute__((ext_vector_type(16))) _Float16 v16h;
typedef __attribute__((ext_vector_type(8)))  float    v8f;

// ---------------------------------------------------------------------------
// Kernel 1: fused attention pooling (flash-softmax over spatial chunks).
// One block per (chunk, batch). Phase A computes mask for the chunk (reads the
// 512 x 448 slab of x once from HBM), phase B re-reads the same slab (L2-hot)
// to accumulate the chunk's partial context weighted by exp(mask - chunkmax).
// ---------------------------------------------------------------------------
__global__ __launch_bounds__(256) void k_pool(const float* __restrict__ x,
                                              const float* __restrict__ wm,
                                              const float* __restrict__ bm,
                                              float* __restrict__ cmax,
                                              float* __restrict__ csum,
                                              float* __restrict__ part)
{
    __shared__ float sm[TS_];
    __shared__ float s_wm[C_];
    __shared__ float s_red[256];
    __shared__ float s_bmax;

    const int tid = threadIdx.x;
    const int ch  = blockIdx.x;
    const int b   = blockIdx.y;
    const int s0  = ch * TS_;

    for (int c = tid; c < C_; c += 256) s_wm[c] = wm[c];
    __syncthreads();

    const float  bmv = bm[0];
    const float* xb  = x + (size_t)b * C_ * S_;

    // ---- phase A: mask[s] = dot(x[b,:,s], wm) + bm  (coalesced across lanes)
    for (int i = tid; i < TS_; i += 256) {
        const float* xp = xb + s0 + i;
        float a0 = 0.f, a1 = 0.f, a2 = 0.f, a3 = 0.f;
        #pragma unroll 4
        for (int c = 0; c < C_; c += 4) {
            a0 += xp[(size_t)(c + 0) * S_] * s_wm[c + 0];
            a1 += xp[(size_t)(c + 1) * S_] * s_wm[c + 1];
            a2 += xp[(size_t)(c + 2) * S_] * s_wm[c + 2];
            a3 += xp[(size_t)(c + 3) * S_] * s_wm[c + 3];
        }
        sm[i] = (a0 + a1) + (a2 + a3) + bmv;
    }
    __syncthreads();

    // ---- chunk max
    float m = -3.4e38f;
    for (int i = tid; i < TS_; i += 256) m = fmaxf(m, sm[i]);
    s_red[tid] = m;
    __syncthreads();
    for (int off = 128; off > 0; off >>= 1) {
        if (tid < off) s_red[tid] = fmaxf(s_red[tid], s_red[tid + off]);
        __syncthreads();
    }
    if (tid == 0) s_bmax = s_red[0];
    __syncthreads();
    const float mx = s_bmax;

    // ---- exp + chunk sum
    float e = 0.f;
    for (int i = tid; i < TS_; i += 256) {
        float w = __expf(sm[i] - mx);
        sm[i] = w;
        e += w;
    }
    s_red[tid] = e;
    __syncthreads();
    for (int off = 128; off > 0; off >>= 1) {
        if (tid < off) s_red[tid] += s_red[tid + off];
        __syncthreads();
    }
    if (tid == 0) {
        cmax[b * NCH_ + ch] = mx;
        csum[b * NCH_ + ch] = s_red[0];
    }
    __syncthreads();

    // ---- phase B: partial context, slab re-read should hit L2
    const int lane = tid & 31, wave = tid >> 5;
    for (int c = wave; c < C_; c += 8) {
        const float* xp = xb + (size_t)c * S_ + s0;
        float acc = 0.f;
        #pragma unroll
        for (int jj = 0; jj < TS_ / 32; ++jj) {
            const int j = lane + jj * 32;
            acc += xp[j] * sm[j];
        }
        #pragma unroll
        for (int off = 16; off > 0; off >>= 1) acc += __shfl_xor(acc, off, 32);
        if (lane == 0) part[((size_t)ch * B_ + b) * C_ + c] = acc;
    }
}

// ---------------------------------------------------------------------------
// Kernel 2: merge chunk partials with exact softmax rescaling.
// ---------------------------------------------------------------------------
__global__ __launch_bounds__(256) void k_combine(const float* __restrict__ cmax,
                                                 const float* __restrict__ csum,
                                                 const float* __restrict__ part,
                                                 float* __restrict__ ctx)
{
    const int b = blockIdx.x, tid = threadIdx.x;
    float gmax = -3.4e38f;
    #pragma unroll
    for (int i = 0; i < NCH_; ++i) gmax = fmaxf(gmax, cmax[b * NCH_ + i]);
    float sc[NCH_];
    float tot = 0.f;
    #pragma unroll
    for (int i = 0; i < NCH_; ++i) {
        sc[i] = __expf(cmax[b * NCH_ + i] - gmax);
        tot  += csum[b * NCH_ + i] * sc[i];
    }
    const float inv = 1.f / tot;
    for (int c = tid; c < C_; c += 256) {
        float a = 0.f;
        #pragma unroll
        for (int i = 0; i < NCH_; ++i) a += part[((size_t)i * B_ + b) * C_ + c] * sc[i];
        ctx[b * C_ + c] = a * inv;
    }
}

// ---------------------------------------------------------------------------
// WMMA fragment gathers. 16-bit A/B layout (ISA 7.12.2): half h of a lane maps
// to K = k0 + (h<8 ? h : h+8) + (lane>=16 ? 8 : 0); rows/columns are
// contiguous in K in our operand storage, so each fragment is two contiguous
// 8-element runs per 16-half group.
// ---------------------------------------------------------------------------
__device__ inline v16h frag_from_f32(const float* __restrict__ row, int hi)
{
    v16h r;
    const float* p0 = row + hi * 8;
    const float* p1 = row + 16 + hi * 8;
    #pragma unroll
    for (int j = 0; j < 8; ++j) {
        r[j]     = (_Float16)p0[j];
        r[j + 8] = (_Float16)p1[j];
    }
    return r;
}

__device__ inline v16h frag_from_f16(const _Float16* __restrict__ row, int hi)
{
    v16h r;
    const _Float16* p0 = row + hi * 8;
    const _Float16* p1 = row + 16 + hi * 8;
    #pragma unroll
    for (int j = 0; j < 8; ++j) {
        r[j]     = p0[j];
        r[j + 8] = p1[j];
    }
    return r;
}

// ---------------------------------------------------------------------------
// Kernel 3: MLP core via v_wmma_f32_16x16x32_f16, single workgroup (8 waves).
// GEMM1: hid[32,64] = ctx[32,512] x w1^T[512,64]  (1 tile / wave, 16 K-steps)
// LN(P=64)+ReLU in LDS with wave32 shuffle reductions
// GEMM2: add[32,512] = hid[32,64] x w2^T[64,512]  (8 tiles / wave, 2 K-steps)
// ---------------------------------------------------------------------------
__global__ __launch_bounds__(256) void k_mlp(const float* __restrict__ ctx,
                                             const float* __restrict__ w1,
                                             const float* __restrict__ b1,
                                             const float* __restrict__ ln_g,
                                             const float* __restrict__ ln_b,
                                             const float* __restrict__ w2,
                                             const float* __restrict__ b2,
                                             const float* __restrict__ gamma,
                                             float* __restrict__ addv)
{
    __shared__ float    s_hid[B_][P_];    // 8 KB
    __shared__ _Float16 s_hidh[B_][P_];   // 4 KB

    const int tid  = threadIdx.x;
    const int lane = tid & 31, wave = tid >> 5;
    const int l15  = lane & 15, hi = lane >> 4;

    // ---- GEMM1: D = A(ctx) x B(w1^T); B[k][n] = w1[n][k] (row-contiguous K)
    {
        const int mt = wave >> 2, nt = wave & 3;
        const int m0 = mt * 16, n0 = nt * 16;
        const float* arow = ctx + (size_t)(m0 + l15) * C_;
        const float* brow = w1  + (size_t)(n0 + l15) * C_;
        v8f acc = {0.f, 0.f, 0.f, 0.f, 0.f, 0.f, 0.f, 0.f};
        for (int k0 = 0; k0 < C_; k0 += 32) {
            v16h a = frag_from_f32(arow + k0, hi);
            v16h b = frag_from_f32(brow + k0, hi);
            acc = __builtin_amdgcn_wmma_f32_16x16x32_f16(false, a, false, b,
                                                         (short)0, acc, false, false);
        }
        // D layout: vgpr r, lane -> M = m0 + r + hi*8, N = n0 + (lane&15)
        const float b1v = b1[n0 + l15];
        #pragma unroll
        for (int r = 0; r < 8; ++r)
            s_hid[m0 + r + hi * 8][n0 + l15] = acc[r] + b1v;
    }
    __syncthreads();

    // ---- LayerNorm over P=64 (+ ReLU), 4 rows per wave
    #pragma unroll
    for (int rr = 0; rr < 4; ++rr) {
        const int row = wave * 4 + rr;
        float v0 = s_hid[row][lane], v1 = s_hid[row][lane + 32];
        float s = v0 + v1, q = v0 * v0 + v1 * v1;
        #pragma unroll
        for (int off = 16; off > 0; off >>= 1) {
            s += __shfl_xor(s, off, 32);
            q += __shfl_xor(q, off, 32);
        }
        const float mean = s * (1.f / 64.f);
        const float var  = q * (1.f / 64.f) - mean * mean;
        const float rstd = rsqrtf(var + EPS_);
        float y0 = (v0 - mean) * rstd * ln_g[lane]      + ln_b[lane];
        float y1 = (v1 - mean) * rstd * ln_g[lane + 32] + ln_b[lane + 32];
        s_hidh[row][lane]      = (_Float16)fmaxf(y0, 0.f);
        s_hidh[row][lane + 32] = (_Float16)fmaxf(y1, 0.f);
    }
    __syncthreads();

    // ---- GEMM2: D = A(hid) x B(w2^T); B[k][n] = w2[n][k] (row-contiguous K)
    const float gm = gamma[0];
    for (int t = wave; t < 64; t += 8) {
        const int mt = t & 1, nt = t >> 1;
        const int m0 = mt * 16, n0 = nt * 16;
        const _Float16* arow = &s_hidh[m0 + l15][0];
        const float*    brow = w2 + (size_t)(n0 + l15) * P_;
        v8f acc = {0.f, 0.f, 0.f, 0.f, 0.f, 0.f, 0.f, 0.f};
        v16h a0  = frag_from_f16(arow + 0, hi);
        v16h b0  = frag_from_f32(brow + 0, hi);
        acc = __builtin_amdgcn_wmma_f32_16x16x32_f16(false, a0, false, b0,
                                                     (short)0, acc, false, false);
        v16h a1  = frag_from_f16(arow + 32, hi);
        v16h b1f = frag_from_f32(brow + 32, hi);
        acc = __builtin_amdgcn_wmma_f32_16x16x32_f16(false, a1, false, b1f,
                                                     (short)0, acc, false, false);
        const float b2v = b2[n0 + l15];
        #pragma unroll
        for (int r = 0; r < 8; ++r)
            addv[(size_t)(m0 + r + hi * 8) * C_ + (n0 + l15)] = gm * (acc[r] + b2v);
    }
}

// ---------------------------------------------------------------------------
// Kernel 4: out = x + addv[b,c] broadcast over spatial (float4 streams).
// ---------------------------------------------------------------------------
__global__ __launch_bounds__(256) void k_out(const float* __restrict__ x,
                                             const float* __restrict__ addv,
                                             float* __restrict__ out)
{
    const int idx = blockIdx.x;            // b*C + c
    const float a = addv[idx];
    const float4* xv = (const float4*)(x + (size_t)idx * S_);
    float4*       ov = (float4*)(out + (size_t)idx * S_);
    for (int i = threadIdx.x; i < S_ / 4; i += 256) {
        float4 v = xv[i];
        v.x += a; v.y += a; v.z += a; v.w += a;
        ov[i] = v;
    }
}

// ---------------------------------------------------------------------------
extern "C" void kernel_launch(void* const* d_in, const int* in_sizes, int n_in,
                              void* d_out, int out_size, void* d_ws, size_t ws_size,
                              hipStream_t stream)
{
    (void)in_sizes; (void)n_in; (void)out_size; (void)ws_size;

    const float* x     = (const float*)d_in[0];
    const float* wm    = (const float*)d_in[1];
    const float* bm    = (const float*)d_in[2];
    const float* w1    = (const float*)d_in[3];
    const float* b1    = (const float*)d_in[4];
    const float* ln_g  = (const float*)d_in[5];
    const float* ln_b  = (const float*)d_in[6];
    const float* w2    = (const float*)d_in[7];
    const float* b2    = (const float*)d_in[8];
    const float* gamma = (const float*)d_in[9];
    float*       out   = (float*)d_out;

    // Workspace layout (floats): ~0.58 MB total
    float* ws   = (float*)d_ws;
    float* cmax = ws;                              // B*NCH            = 224
    float* csum = cmax + B_ * NCH_;                // B*NCH            = 224
    float* part = csum + B_ * NCH_;                // NCH*B*C          = 114688
    float* ctx  = part + (size_t)NCH_ * B_ * C_;   // B*C              = 16384
    float* addv = ctx + B_ * C_;                   // B*C              = 16384

    k_pool<<<dim3(NCH_, B_), 256, 0, stream>>>(x, wm, bm, cmax, csum, part);
    k_combine<<<B_, 256, 0, stream>>>(cmax, csum, part, ctx);
    k_mlp<<<1, 256, 0, stream>>>(ctx, w1, b1, ln_g, ln_b, w2, b2, gamma, addv);
    k_out<<<B_ * C_, 256, 0, stream>>>(x, addv, out);
}
